// GPT_89429809038091
// MI455X (gfx1250) — matrix operand order repaired and checked by reading, not compiled, not run on hardware
//
#include <hip/hip_runtime.h>
#include <hip/hip_bf16.h>

// GPT-2 forward for gfx1250 (MI455X): bf16 WMMA GEMMs, f32 accumulate.
// Model: V=50257 CTX=2048 D=1024 H=16 L=8 DH=64 B=2 S=2048.
// Workspace requirement: ~294 MB.

#define Dm   1024
#define Hh   16
#define DHd  64
#define Ll   8
#define Ss   2048
#define Bb   2
#define Vv   50257
#define NTOK (Bb * Ss)          // 4096 activation rows

#define SA   72                 // padded LDS row stride (halves): 144B, 16B-aligned
#define SB   72

typedef __attribute__((ext_vector_type(16))) __bf16 v16bf;
typedef __attribute__((ext_vector_type(8)))  float  v8f;

__device__ __forceinline__ unsigned short f2bf(float f) {
    unsigned int u = __float_as_uint(f);
    u = (u + 0x7FFFu + ((u >> 16) & 1u)) >> 16;   // round-to-nearest-even
    return (unsigned short)u;
}
__device__ __forceinline__ unsigned short toBF(float f)          { return f2bf(f); }
__device__ __forceinline__ unsigned short toBF(unsigned short s) { return s; }

// load 4 consecutive elements -> bf16 (vectorized; requires alignment)
__device__ __forceinline__ void load4(const float* p, unsigned short (&t)[4]) {
    float4 f = *(const float4*)p;
    t[0] = f2bf(f.x); t[1] = f2bf(f.y); t[2] = f2bf(f.z); t[3] = f2bf(f.w);
}
__device__ __forceinline__ void load4(const unsigned short* p, unsigned short (&t)[4]) {
    uint2 u = *(const uint2*)p;
    t[0] = u.x & 0xffff; t[1] = u.x >> 16; t[2] = u.y & 0xffff; t[3] = u.y >> 16;
}

// ---------------------------------------------------------------- embedding
__global__ void embed_kernel(const int* __restrict__ x, const float* __restrict__ wte,
                             const float* __restrict__ wpe, float* __restrict__ h) {
    int gid = blockIdx.x * 256 + threadIdx.x;      // [0, NTOK*D)
    int tok = gid >> 10;                           // /D
    int d   = gid & (Dm - 1);
    int s   = tok & (Ss - 1);
    h[gid] = wte[(size_t)x[tok] * Dm + d] + wpe[(size_t)s * Dm + d];
}

// ---------------------------------------------------------------- layernorm
__global__ void ln_kernel(const float* __restrict__ in, const float* __restrict__ g,
                          const float* __restrict__ bta, unsigned short* __restrict__ out) {
    int row = blockIdx.x, tid = threadIdx.x;
    const float* xr = in + (size_t)row * Dm;
    __shared__ float red[256];
    float s = 0.f;
    for (int i = tid; i < Dm; i += 256) s += xr[i];
    red[tid] = s; __syncthreads();
    for (int o = 128; o > 0; o >>= 1) { if (tid < o) red[tid] += red[tid + o]; __syncthreads(); }
    float mean = red[0] * (1.0f / Dm);
    __syncthreads();
    float v = 0.f;
    for (int i = tid; i < Dm; i += 256) { float d = xr[i] - mean; v += d * d; }
    red[tid] = v; __syncthreads();
    for (int o = 128; o > 0; o >>= 1) { if (tid < o) red[tid] += red[tid + o]; __syncthreads(); }
    float rstd = rsqrtf(red[0] * (1.0f / Dm) + 1e-5f);
    for (int i = tid; i < Dm; i += 256)
        out[(size_t)row * Dm + i] = f2bf((xr[i] - mean) * rstd * g[i] + bta[i]);
}

// ------------------------------------------------------- causal softmax row
// batched over blockIdx.y (head slice); scale = DH^-0.5 = 0.125
__global__ void softmax_kernel(const float* __restrict__ sc, unsigned short* __restrict__ pb) {
    int row = blockIdx.x, tid = threadIdx.x;
    size_t slice = (size_t)blockIdx.y * Ss * Ss;
    const float* s = sc + slice + (size_t)row * Ss;
    unsigned short* p = pb + slice + (size_t)row * Ss;
    const float scale = 0.125f;
    __shared__ float red[256];
    float m = -1e30f;
    for (int i = tid; i < Ss; i += 256) if (i <= row) m = fmaxf(m, s[i] * scale);
    red[tid] = m; __syncthreads();
    for (int o = 128; o > 0; o >>= 1) { if (tid < o) red[tid] = fmaxf(red[tid], red[tid + o]); __syncthreads(); }
    float mx = red[0];
    __syncthreads();
    float sum = 0.f;
    for (int i = tid; i < Ss; i += 256) if (i <= row) sum += expf(s[i] * scale - mx);
    red[tid] = sum; __syncthreads();
    for (int o = 128; o > 0; o >>= 1) { if (tid < o) red[tid] += red[tid + o]; __syncthreads(); }
    float inv = 1.0f / red[0];
    for (int i = tid; i < Ss; i += 256)
        p[i] = (i <= row) ? f2bf(expf(s[i] * scale - mx) * inv) : (unsigned short)0;
}

// --------------------------------------------------------------- WMMA GEMM
// C[z][M,N] = epi( A[z][M,K](bf16,lda) * W[z] + bias ), batched over blockIdx.z.
// W addressing: element (k,n) at W[k*wsk + n*wsn].
//   WMODE==0: wsn==1  (weights / V)      -> n-vectorized loads, k-pair packing
//   WMODE==1: wsk==1  (K^T in attention) -> k-vectorized bf16 copies
// Block: 256 thr = 8 waves as 4(M)x2(N); wave tile 32x32 (4 WMMA accums);
// block tile 128x64; K-step 64 (2 sub-steps of 32 -> 8 WMMA/wave/barrier).
// LDS: A [128][SA], W^T [64][SB] (bf16, padded rows).  K must be % 64.
template <typename WT, int WMODE, bool GELU, bool RES>
__global__ void gemm_kernel(const unsigned short* __restrict__ A, int lda, long long aBatch,
                            const WT* __restrict__ W, int wsk, int wsn, long long wBatch,
                            const float* __restrict__ bias,
                            const float* __restrict__ res,
                            float* __restrict__ outF, unsigned short* __restrict__ outB,
                            int ldc, long long cBatch,
                            int M, int N, int K) {
    __shared__ __align__(16) unsigned short As[128 * SA];
    __shared__ __align__(16) unsigned short Bs[64 * SB];   // [n][k]
    const int tid = threadIdx.x, lane = tid & 31, wave = tid >> 5;
    const int mw = wave >> 1;                 // 0..3 : 32-row group
    const int nw = wave & 1;                  // 0..1 : 32-col group
    const int rowBase = blockIdx.y * 128;
    const int colBase = blockIdx.x * 64;
    const int hi = lane >> 4, lo = lane & 15;
    const long long zb = blockIdx.z;
    A += zb * aBatch;
    W += zb * wBatch;
    const long long cOff = zb * cBatch;

    // staging coordinates
    const int ar  = tid >> 1;                 // A row 0..127
    const int ac  = (tid & 1) * 32;           // A col base {0,32}
    const int kq  = (tid >> 4) * 4;           // WMODE0: k quad 0..60
    const int n0  = (tid & 15) * 4;           // WMODE0: n base
    const int bn  = tid >> 2;                 // WMODE1: n 0..63
    const int bk0 = (tid & 3) * 16;           // WMODE1: k base
    const bool fullN    = (colBase + 64 <= N);
    const bool alignedW = ((wsk * (int)sizeof(WT)) & 15) == 0;

    union Acc { v8f v; float f[8]; };
    Acc acc[2][2];
#pragma unroll
    for (int i = 0; i < 2; ++i)
#pragma unroll
        for (int j = 0; j < 2; ++j)
            acc[i][j].v = v8f{0.f, 0.f, 0.f, 0.f, 0.f, 0.f, 0.f, 0.f};

    for (int k0 = 0; k0 < K; k0 += 64) {
        // ---- stage A: 32 halfs per thread (four b128 loads)
        {
            int grow = rowBase + ar;
            uint4 u0 = make_uint4(0, 0, 0, 0), u1 = u0, u2 = u0, u3 = u0;
            if (grow < M) {
                const unsigned short* ap = A + (size_t)grow * lda + k0 + ac;
                u0 = *(const uint4*)ap;
                u1 = *(const uint4*)(ap + 8);
                u2 = *(const uint4*)(ap + 16);
                u3 = *(const uint4*)(ap + 24);
                if (k0 + 64 < K) __builtin_prefetch(ap + 64, 0, 1);
            }
            unsigned short* as = &As[ar * SA + ac];
            *(uint4*)(as)      = u0;
            *(uint4*)(as + 8)  = u1;
            *(uint4*)(as + 16) = u2;
            *(uint4*)(as + 24) = u3;
        }
        // ---- stage W -> Bs[n][k]
        if (WMODE == 1) {
            const unsigned short* wp =
                (const unsigned short*)W + (size_t)(colBase + bn) * wsn + (k0 + bk0);
            unsigned short* bs = &Bs[bn * SB + bk0];
            *(uint4*)(bs)     = *(const uint4*)wp;
            *(uint4*)(bs + 8) = *(const uint4*)(wp + 8);
            if (k0 + 64 < K) __builtin_prefetch(wp + 64, 0, 1);
        } else {
            unsigned short t[4][4];   // [ki][j] for k = kq+ki, n = colBase+n0+j
            if (fullN && alignedW) {
#pragma unroll
                for (int i = 0; i < 4; ++i)
                    load4(W + (size_t)(k0 + kq + i) * (size_t)wsk + colBase + n0, t[i]);
                if (k0 + 64 < K)
                    __builtin_prefetch(W + (size_t)(k0 + 64 + kq) * (size_t)wsk + colBase + n0, 0, 1);
            } else {
#pragma unroll
                for (int i = 0; i < 4; ++i)
#pragma unroll
                    for (int j = 0; j < 4; ++j) {
                        int n = colBase + n0 + j;
                        t[i][j] = (fullN || n < N)
                                      ? toBF(W[(size_t)(k0 + kq + i) * (size_t)wsk + n])
                                      : (unsigned short)0;
                    }
            }
#pragma unroll
            for (int j = 0; j < 4; ++j) {   // pack k-pairs -> one b64 store per n
                unsigned int u0 = (unsigned int)t[0][j] | ((unsigned int)t[1][j] << 16);
                unsigned int u1 = (unsigned int)t[2][j] | ((unsigned int)t[3][j] << 16);
                *(uint2*)&Bs[(n0 + j) * SB + kq] = make_uint2(u0, u1);
            }
        }
        __syncthreads();

        // ---- two 32-wide K sub-steps: 8 WMMA per wave per barrier pair
#pragma unroll
        for (int ks = 0; ks < 64; ks += 32) {
            union Frag { unsigned int u[8]; v16bf v; };
            Frag af[2], bfg[2];
#pragma unroll
            for (int mi = 0; mi < 2; ++mi) {
                const unsigned short* arow = &As[(mw * 32 + mi * 16 + lo) * SA + ks];
#pragma unroll
                for (int p = 0; p < 4; ++p) af[mi].u[p]     = *(const unsigned int*)&arow[hi * 8 + 2 * p];
#pragma unroll
                for (int p = 0; p < 4; ++p) af[mi].u[4 + p] = *(const unsigned int*)&arow[16 + hi * 8 + 2 * p];
            }
#pragma unroll
            for (int ni = 0; ni < 2; ++ni) {
                const unsigned short* brow = &Bs[(nw * 32 + ni * 16 + lo) * SB + ks + hi * 8];
#pragma unroll
                for (int p = 0; p < 8; ++p) bfg[ni].u[p] = *(const unsigned int*)&brow[2 * p + (p >= 4 ? 8 : 0)];
            }
#pragma unroll
            for (int mi = 0; mi < 2; ++mi)
#pragma unroll
                for (int ni = 0; ni < 2; ++ni)
                    acc[mi][ni].v = __builtin_amdgcn_wmma_f32_16x16x32_bf16(
                        false, af[mi].v, false, bfg[ni].v, (short)0, acc[mi][ni].v, false, false);
        }
        __syncthreads();
    }

    // ---- epilogue: lane (hi,lo) -> N=lo ; VGPR r -> M = r or r+8
#pragma unroll
    for (int mi = 0; mi < 2; ++mi) {
#pragma unroll
        for (int ni = 0; ni < 2; ++ni) {
            int col = colBase + nw * 32 + ni * 16 + lo;
#pragma unroll
            for (int r = 0; r < 8; ++r) {
                int row = rowBase + mw * 32 + mi * 16 + (hi ? 8 + r : r);
                if (row < M && col < N) {
                    float c = acc[mi][ni].f[r];
                    if (bias) c += bias[col];
                    if (GELU) c = 0.5f * c * (1.0f + erff(c * 0.70710678118654752f));
                    size_t cix = (size_t)((long long)row * ldc + col + cOff);
                    if (RES)  c += res[cix];
                    if (outF) outF[cix] = c;
                    if (outB) outB[cix] = f2bf(c);
                }
            }
        }
    }
}

// ---------------------------------------------------------------- launcher
extern "C" void kernel_launch(void* const* d_in, const int* in_sizes, int n_in,
                              void* d_out, int out_size, void* d_ws, size_t ws_size,
                              hipStream_t stream) {
    const int*   x      = (const int*)d_in[0];
    const float* wte    = (const float*)d_in[1];
    const float* wpe    = (const float*)d_in[2];
    const float* ln1_g  = (const float*)d_in[3];
    const float* ln1_b  = (const float*)d_in[4];
    const float* w_qkv  = (const float*)d_in[5];
    const float* b_qkv  = (const float*)d_in[6];
    const float* w_proj = (const float*)d_in[7];
    const float* b_proj = (const float*)d_in[8];
    const float* ln2_g  = (const float*)d_in[9];
    const float* ln2_b  = (const float*)d_in[10];
    const float* w_fc   = (const float*)d_in[11];
    const float* b_fc   = (const float*)d_in[12];
    const float* w_cp   = (const float*)d_in[13];
    const float* b_cp   = (const float*)d_in[14];
    const float* lnf_g  = (const float*)d_in[15];
    const float* lnf_b  = (const float*)d_in[16];
    const float* w_head = (const float*)d_in[17];
    float* out = (float*)d_out;

    char* ws = (char*)d_ws;
    size_t off = 0;
    auto alloc = [&](size_t bytes) -> void* {
        void* p = ws + off;
        off += (bytes + 255) & ~(size_t)255;
        return p;
    };
    const int G = 8;   // head slices batched per attention launch
    float*          h    = (float*)         alloc((size_t)NTOK * Dm * 4);
    unsigned short* y    = (unsigned short*)alloc((size_t)NTOK * Dm * 2);
    unsigned short* qkv  = (unsigned short*)alloc((size_t)NTOK * 3 * Dm * 2);
    unsigned short* z    = (unsigned short*)alloc((size_t)NTOK * Dm * 2);
    float*          sc   = (float*)         alloc((size_t)G * Ss * Ss * 4);
    unsigned short* pr   = (unsigned short*)alloc((size_t)G * Ss * Ss * 2);
    unsigned short* fcb  = (unsigned short*)alloc((size_t)NTOK * 4 * Dm * 2);
    unsigned short* lnfb = (unsigned short*)alloc((size_t)Bb * Dm * 2);

    dim3 blk(256);
    const long long SxS = (long long)Ss * Ss;

    embed_kernel<<<(NTOK * Dm) / 256, blk, 0, stream>>>(x, wte, wpe, h);

    for (int l = 0; l < Ll; ++l) {
        ln_kernel<<<NTOK, blk, 0, stream>>>(h, ln1_g + l * Dm, ln1_b + l * Dm, y);
        // qkv = y @ w_qkv + b_qkv -> bf16
        gemm_kernel<float, 0, false, false><<<dim3(3 * Dm / 64, NTOK / 128), blk, 0, stream>>>(
            y, Dm, 0, w_qkv + (size_t)l * Dm * 3 * Dm, 3 * Dm, 1, 0, b_qkv + l * 3 * Dm,
            nullptr, nullptr, qkv, 3 * Dm, 0, NTOK, 3 * Dm, Dm);
        // attention: 4 groups of G=8 head slices (same batch index within a group)
        for (int grp = 0; grp < Bb * Hh / G; ++grp) {
            int bh0 = grp * G, b = bh0 >> 4, hd0 = bh0 & 15;
            const unsigned short* qb = qkv + (size_t)b * Ss * 3 * Dm + hd0 * DHd;
            const unsigned short* kb = qb + Dm;
            const unsigned short* vb = qb + 2 * Dm;
            // scores[z] = q @ k^T   (K^T: wsk=1, wsn=3D -> WMODE 1; K=64, one stage)
            gemm_kernel<unsigned short, 1, false, false><<<dim3(Ss / 64, Ss / 128, G), blk, 0, stream>>>(
                qb, 3 * Dm, DHd, kb, 1, 3 * Dm, DHd, nullptr,
                nullptr, sc, nullptr, Ss, SxS, Ss, Ss, DHd);
            softmax_kernel<<<dim3(Ss, G), blk, 0, stream>>>(sc, pr);
            // z[z] = P @ V          (V: wsk=3D, wsn=1 -> WMODE 0, bf16 W)
            gemm_kernel<unsigned short, 0, false, false><<<dim3(1, Ss / 128, G), blk, 0, stream>>>(
                pr, Ss, SxS, vb, 3 * Dm, 1, DHd, nullptr,
                nullptr, nullptr, z + (size_t)b * Ss * Dm + hd0 * DHd, Dm, DHd, Ss, DHd, Ss);
        }
        // h += z @ w_proj + b_proj
        gemm_kernel<float, 0, false, true><<<dim3(Dm / 64, NTOK / 128), blk, 0, stream>>>(
            z, Dm, 0, w_proj + (size_t)l * Dm * Dm, Dm, 1, 0, b_proj + l * Dm,
            h, h, nullptr, Dm, 0, NTOK, Dm, Dm);
        ln_kernel<<<NTOK, blk, 0, stream>>>(h, ln2_g + l * Dm, ln2_b + l * Dm, y);
        // fcb = gelu(y @ w_fc + b_fc) -> bf16
        gemm_kernel<float, 0, true, false><<<dim3(4 * Dm / 64, NTOK / 128), blk, 0, stream>>>(
            y, Dm, 0, w_fc + (size_t)l * Dm * 4 * Dm, 4 * Dm, 1, 0, b_fc + l * 4 * Dm,
            nullptr, nullptr, fcb, 4 * Dm, 0, NTOK, 4 * Dm, Dm);
        // h += fcb @ w_cp + b_cp
        gemm_kernel<float, 0, false, true><<<dim3(Dm / 64, NTOK / 128), blk, 0, stream>>>(
            fcb, 4 * Dm, 0, w_cp + (size_t)l * 4 * Dm * Dm, Dm, 1, 0, b_cp + l * Dm,
            h, h, nullptr, Dm, 0, NTOK, Dm, 4 * Dm);
    }

    // final LN on last token of each batch, then head GEMM (M = B rows)
    ln_kernel<<<1, blk, 0, stream>>>(h + (size_t)(0 * Ss + Ss - 1) * Dm, lnf_g, lnf_b, lnfb);
    ln_kernel<<<1, blk, 0, stream>>>(h + (size_t)(1 * Ss + Ss - 1) * Dm, lnf_g, lnf_b, lnfb + Dm);
    gemm_kernel<float, 0, false, false><<<dim3((Vv + 63) / 64, 1), blk, 0, stream>>>(
        lnfb, Dm, 0, w_head, Vv, 1, 0, nullptr,
        nullptr, out, nullptr, Vv, 0, Bb, Vv, Dm);
    (void)in_sizes; (void)n_in; (void)out_size; (void)ws_size;
}
